// DecoderBlock_13176959664293
// MI455X (gfx1250) — matrix operand order repaired and checked by reading, not compiled, not run on hardware
//
#include <hip/hip_runtime.h>

// ---------------------------------------------------------------------------
// CDNA5 (gfx1250) AFT-local decoder block.
// Dense GEMMs: workgroup-cooperative bf16 WMMA with async global->LDS B-tiles.
// ---------------------------------------------------------------------------

typedef __attribute__((ext_vector_type(16))) __bf16 v16bf;
typedef __attribute__((ext_vector_type(8)))  float  v8f;
typedef __attribute__((address_space(3))) unsigned short lds_ushort;

union FragBF {
    v16bf v;
    unsigned short u[16];
    uint4 q[2];
};

__device__ __forceinline__ unsigned short cvt_bf16(float f) {
    unsigned int u = __float_as_uint(f);
    if ((u & 0x7fffffffu) > 0x7f800000u) return (unsigned short)((u >> 16) | 0x0040u); // quiet NaN
    return (unsigned short)((u + 0x7fffu + ((u >> 16) & 1u)) >> 16);                   // RNE
}
__device__ __forceinline__ float bf16_to_f32(unsigned short u) {
    return __uint_as_float(((unsigned int)u) << 16);
}
__device__ __forceinline__ uint4 ld16(const unsigned short* p) {
    return *reinterpret_cast<const uint4*>(p);
}

#define WMMA_BF16(ACC, AF, BF) \
    ACC = __builtin_amdgcn_wmma_f32_16x16x32_bf16(false, (AF).v, false, (BF).v, (short)0, ACC, false, false)

// 16B async copy global -> LDS, tracked by ASYNCcnt (ISA 15.18.3 op 98)
#define ASYNC_CP_B128(LDSDST, GSRC) \
    asm volatile("global_load_async_to_lds_b128 %0, %1, off" :: "v"(LDSDST), "v"(GSRC) : "memory")
#define WAIT_ASYNC0() asm volatile("s_wait_asynccnt 0x0" ::: "memory")

// ---------------------------------------------------------------------------
// 1) emb = x + pos_encod (f32) + bf16 copy
// ---------------------------------------------------------------------------
__global__ __launch_bounds__(256) void add_embed_kernel(
    const float* __restrict__ x, const float* __restrict__ pe,
    float* __restrict__ emb, unsigned short* __restrict__ embh, size_t n)
{
    size_t i = (size_t)blockIdx.x * blockDim.x + threadIdx.x;
    if (i < n) {
        float e = x[i] + pe[i];
        emb[i]  = e;
        embh[i] = cvt_bf16(e);
    }
}

// ---------------------------------------------------------------------------
// 2) f32 [R,C] -> bf16 [C,R]  (weights pre-transposed: WMMA B-fragments become
//    k-contiguous rows, read as two 8-half chunks per lane)
// ---------------------------------------------------------------------------
__global__ __launch_bounds__(256) void transpose_cvt_kernel(
    const float* __restrict__ src, unsigned short* __restrict__ dst, int R, int C)
{
    __shared__ float tile[32][33];
    int c0 = blockIdx.x * 32, r0 = blockIdx.y * 32;
    int tx = threadIdx.x, ty = threadIdx.y;               // (32,8)
    for (int i = 0; i < 32; i += 8)
        tile[ty + i][tx] = src[(size_t)(r0 + ty + i) * C + c0 + tx];
    __syncthreads();
    for (int i = 0; i < 32; i += 8)
        dst[(size_t)(c0 + ty + i) * R + r0 + tx] = cvt_bf16(tile[tx][ty + i]);
}

// ---------------------------------------------------------------------------
// 3) Cooperative bf16 WMMA GEMM:  C[M,N] = A[M,K] @ BT[N,K]^T + bias.
//    Workgroup (8 waves) -> 128x64 tile; wave w -> rows m0+16w.
//    B k-slab (64 rows x 32 halves = 4KB) staged in LDS via async copies,
//    double buffered; A fragment register-pipelined.
// ---------------------------------------------------------------------------
__global__ __launch_bounds__(256) void gemm_bf16_coop_kernel(
    const unsigned short* __restrict__ A, int lda,
    const unsigned short* __restrict__ BT, int ldb,
    float* __restrict__ Cf, unsigned short* __restrict__ Ch, int ldc,
    const float* __restrict__ bias, int M, int N, int K, int relu)
{
    __shared__ unsigned short btile[2][64 * 32];          // 2 x 4KB
    int lane = threadIdx.x & 31;
    int wave = threadIdx.x >> 5;
    int wgRows = M >> 7, wgCols = N >> 6;
    int wg = blockIdx.x;
    if (wg >= wgRows * wgCols) return;
    int trg = wg % wgRows, tcg = wg / wgRows;             // row-fastest: neighbors share B
    int m0 = (trg << 7) + (wave << 4);
    int n0 = tcg << 6;
    int nl = lane & 15, hi = lane >> 4;

    // cooperative B loader: each of 256 threads owns one 16B chunk of the slab
    int brow = threadIdx.x >> 2;                          // 0..63  (BT row = output col)
    int bch  = threadIdx.x & 3;                           // 0..3   (k chunk of 8 halves)
    const unsigned short* bsrc = BT + (size_t)(n0 + brow) * ldb + bch * 8;
    lds_ushort* bdst0 = (lds_ushort*)&btile[0][brow * 32 + bch * 8];
    lds_ushort* bdst1 = (lds_ushort*)&btile[1][brow * 32 + bch * 8];

    ASYNC_CP_B128(bdst0, bsrc);                           // stage k0 = 0

    const unsigned short* arow = A + (size_t)(m0 + nl) * lda;
    FragBF a;
    a.q[0] = ld16(arow + hi * 8);                         // k = hi*8 + 0..7
    a.q[1] = ld16(arow + 16 + hi * 8);                    // k = 16 + hi*8 + 0..7

    v8f acc0 = {}, acc1 = {}, acc2 = {}, acc3 = {};

    WAIT_ASYNC0();
    __syncthreads();

    int cur = 0;
    for (int k0 = 0; k0 < K; k0 += 32) {
        int kn = k0 + 32;
        if (kn < K) {                                     // issue next slab while computing
            if (cur) { ASYNC_CP_B128(bdst0, bsrc + kn); }
            else     { ASYNC_CP_B128(bdst1, bsrc + kn); }
        }
        FragBF an;
        if (kn < K) {
            an.q[0] = ld16(arow + kn + hi * 8);
            an.q[1] = ld16(arow + kn + 16 + hi * 8);
        }
        const unsigned short* lb = &btile[cur][nl * 32 + hi * 8];
        FragBF b;
        b.q[0] = ld16(lb +    0); b.q[1] = ld16(lb +   16); WMMA_BF16(acc0, a, b);
        b.q[0] = ld16(lb +  512); b.q[1] = ld16(lb +  528); WMMA_BF16(acc1, a, b);
        b.q[0] = ld16(lb + 1024); b.q[1] = ld16(lb + 1040); WMMA_BF16(acc2, a, b);
        b.q[0] = ld16(lb + 1536); b.q[1] = ld16(lb + 1552); WMMA_BF16(acc3, a, b);
        WAIT_ASYNC0();                                    // own slab writes landed
        __syncthreads();                                  // everyone done reading cur / writing nxt
        a = an;
        cur ^= 1;
    }

#define GEMM_EPI(ACC, SUB)                                                      \
    {                                                                           \
        int col = n0 + (SUB) * 16 + nl;                                         \
        float bb = bias ? bias[col] : 0.0f;                                     \
        for (int r = 0; r < 8; ++r) {                                           \
            float val = ACC[r] + bb;                                            \
            if (relu) val = fmaxf(val, 0.0f);                                   \
            size_t o = (size_t)(m0 + hi * 8 + r) * ldc + col;                   \
            if (Ch) Ch[o] = cvt_bf16(val); else Cf[o] = val;                    \
        }                                                                       \
    }
    GEMM_EPI(acc0, 0) GEMM_EPI(acc1, 1) GEMM_EPI(acc2, 2) GEMM_EPI(acc3, 3)
#undef GEMM_EPI
}

// ---------------------------------------------------------------------------
// 4) maxk[b,d] = max_t k[b,t,d]
// ---------------------------------------------------------------------------
__global__ __launch_bounds__(256) void colmax_kernel(
    const float* __restrict__ k32, float* __restrict__ maxk, int Bb, int T_, int D_)
{
    int idx = blockIdx.x * blockDim.x + threadIdx.x;
    if (idx >= Bb * D_) return;
    int b = idx / D_, d = idx % D_;
    const float* p = k32 + (size_t)b * T_ * D_ + d;
    float m = -INFINITY;
    for (int t = 0; t < T_; ++t) m = fmaxf(m, p[(size_t)t * D_]);
    maxk[idx] = m;
}

// ---------------------------------------------------------------------------
// 5) ekvT[b, 2D, T] bf16: cols [0,D)=exp(k-maxk)*v, cols [D,2D)=exp(k-maxk)
// ---------------------------------------------------------------------------
__global__ __launch_bounds__(256) void build_ekvT_kernel(
    const float* __restrict__ k32, const float* __restrict__ v32,
    const float* __restrict__ maxk, unsigned short* __restrict__ ekvT, int T_, int D_)
{
    __shared__ float t1[32][33];
    __shared__ float t2[32][33];
    int b = blockIdx.z;
    int t0 = blockIdx.x * 32, d0 = blockIdx.y * 32;
    int tx = threadIdx.x, ty = threadIdx.y;               // (32,8)
    for (int i = 0; i < 32; i += 8) {
        int t = t0 + ty + i, d = d0 + tx;
        size_t off = ((size_t)b * T_ + t) * D_ + d;
        float e = __expf(k32[off] - maxk[b * D_ + d]);
        t1[ty + i][tx] = e * v32[off];
        t2[ty + i][tx] = e;
    }
    __syncthreads();
    unsigned short* base = ekvT + (size_t)b * (2 * D_) * T_;
    for (int i = 0; i < 32; i += 8) {
        int d = d0 + ty + i, t = t0 + tx;
        base[(size_t)d * T_ + t]        = cvt_bf16(t1[tx][ty + i]);
        base[(size_t)(D_ + d) * T_ + t] = cvt_bf16(t2[tx][ty + i]);
    }
}

// ---------------------------------------------------------------------------
// 6) S[b,c] = sum_t ekvT[b,c,t]
// ---------------------------------------------------------------------------
__global__ __launch_bounds__(256) void colsum_kernel(
    const unsigned short* __restrict__ ekvT, float* __restrict__ Ssum, int T_)
{
    int col = blockIdx.x;
    const unsigned short* p = ekvT + (size_t)col * T_;
    float s = 0.0f;
    for (int t = threadIdx.x; t < T_; t += blockDim.x) s += bf16_to_f32(p[t]);
    __shared__ float red[256];
    red[threadIdx.x] = s; __syncthreads();
    for (int st = 128; st > 0; st >>= 1) {
        if (threadIdx.x < st) red[threadIdx.x] += red[threadIdx.x + st];
        __syncthreads();
    }
    if (threadIdx.x == 0) Ssum[col] = red[0];
}

// ---------------------------------------------------------------------------
// 7) Banded ew pack:  A[r,c] = (|t-j|<S) ? exp(pb-mw)-exp(-mw) : 0  (bf16),
//    plus emw[t] = exp(-mw[t]).  Out-of-band handled via column sums later.
// ---------------------------------------------------------------------------
__global__ __launch_bounds__(256) void ew_pack_kernel(
    const float* __restrict__ pb, unsigned short* __restrict__ ewp,
    float* __restrict__ emw, int T_, int S_, int KW, int JSHIFT)
{
    __shared__ float rowmax[16];
    __shared__ float red[256];
    int tile = blockIdx.x, t0 = tile * 16;
    int r = threadIdx.x >> 4, s = threadIdx.x & 15;
    int t = t0 + r;
    int jlo = t - (S_ - 1); if (jlo < 0) jlo = 0;
    int jhi = t + (S_ - 1); if (jhi >= T_) jhi = T_ - 1;
    float m = 0.0f;                                       // zeros outside band always present
    for (int j = jlo + s; j <= jhi; j += 16) m = fmaxf(m, pb[(size_t)t * T_ + j]);
    red[threadIdx.x] = m; __syncthreads();
    for (int st = 8; st > 0; st >>= 1) {
        if (s < st) red[threadIdx.x] = fmaxf(red[threadIdx.x], red[threadIdx.x + st]);
        __syncthreads();
    }
    if (s == 0) { rowmax[r] = red[threadIdx.x]; emw[t] = __expf(-red[threadIdx.x]); }
    __syncthreads();

    int j0 = t0 - JSHIFT; if (j0 < 0) j0 = 0; if (j0 > T_ - KW) j0 = T_ - KW;
    for (int idx = threadIdx.x; idx < 16 * KW; idx += 256) {
        int rr = idx / KW, cc = idx % KW;
        int tt = t0 + rr, j = j0 + cc;
        float mw = rowmax[rr];
        int dd = j - tt; if (dd < 0) dd = -dd;
        float val = (dd < S_) ? (__expf(pb[(size_t)tt * T_ + j] - mw) - __expf(-mw)) : 0.0f;
        ewp[(size_t)tile * 16 * KW + idx] = cvt_bf16(val);
    }
}

// ---------------------------------------------------------------------------
// 8) Banded GEMM: ND[b,t,0:2D] = EW_tile(16xKW) @ ekvT[b,:,j0:j0+KW]^T.
//    All 5 fragments loaded up front per k-step so they clause together.
// ---------------------------------------------------------------------------
__global__ __launch_bounds__(256) void band_gemm_kernel(
    const unsigned short* __restrict__ EW,   // [T/16][16][KW]
    const unsigned short* __restrict__ EKVT, // [B][2D][T]
    float* __restrict__ ND,                  // [B][T][2D]
    int T_, int D2, int KW, int JSHIFT)
{
    int lane = threadIdx.x & 31;
    int wave = threadIdx.x >> 5;
    int gid  = blockIdx.x * (blockDim.x >> 5) + wave;
    int tileRows = T_ >> 4, tileCols = D2 >> 6;
    int perB = tileRows * tileCols;
    if (gid >= 4 * perB) return;
    int b = gid / perB;
    int g = gid % perB;
    int tr = g % tileRows, tc = g / tileRows;
    int t0 = tr << 4, n0 = tc << 6;
    int j0 = t0 - JSHIFT; if (j0 < 0) j0 = 0; if (j0 > T_ - KW) j0 = T_ - KW;
    int nl = lane & 15, hi = lane >> 4;

    v8f acc0 = {}, acc1 = {}, acc2 = {}, acc3 = {};
    const unsigned short* arow = EW + ((size_t)tr * 16 + nl) * KW;
    const unsigned short* bbase = EKVT + (size_t)b * D2 * T_;
    const unsigned short* b0 = bbase + (size_t)(n0 +  0 + nl) * T_ + j0 + hi * 8;
    const unsigned short* b1 = bbase + (size_t)(n0 + 16 + nl) * T_ + j0 + hi * 8;
    const unsigned short* b2 = bbase + (size_t)(n0 + 32 + nl) * T_ + j0 + hi * 8;
    const unsigned short* b3 = bbase + (size_t)(n0 + 48 + nl) * T_ + j0 + hi * 8;

    for (int k0 = 0; k0 < KW; k0 += 32) {
        if (k0 + 64 < KW) {
            __builtin_prefetch(arow + k0 + 64, 0, 3);
            __builtin_prefetch(b0 + k0 + 64, 0, 3);
        }
        FragBF a, f0, f1, f2, f3;
        a.q[0]  = ld16(arow + k0 + hi * 8);
        a.q[1]  = ld16(arow + k0 + 16 + hi * 8);
        f0.q[0] = ld16(b0 + k0); f0.q[1] = ld16(b0 + k0 + 16);
        f1.q[0] = ld16(b1 + k0); f1.q[1] = ld16(b1 + k0 + 16);
        f2.q[0] = ld16(b2 + k0); f2.q[1] = ld16(b2 + k0 + 16);
        f3.q[0] = ld16(b3 + k0); f3.q[1] = ld16(b3 + k0 + 16);
        WMMA_BF16(acc0, a, f0);
        WMMA_BF16(acc1, a, f1);
        WMMA_BF16(acc2, a, f2);
        WMMA_BF16(acc3, a, f3);
    }

    float* crow = ND + (size_t)b * T_ * D2;
#define BAND_EPI(ACC, SUB)                                                      \
    {                                                                           \
        int col = n0 + (SUB) * 16 + nl;                                         \
        for (int r = 0; r < 8; ++r)                                             \
            crow[(size_t)(t0 + hi * 8 + r) * D2 + col] = ACC[r];                \
    }
    BAND_EPI(acc0, 0) BAND_EPI(acc1, 1) BAND_EPI(acc2, 2) BAND_EPI(acc3, 3)
#undef BAND_EPI
}

// ---------------------------------------------------------------------------
// 9) combine: y = sigmoid(q) * (nd_num + emw*S_num) / (nd_den + emw*S_den)
// ---------------------------------------------------------------------------
__global__ __launch_bounds__(256) void combine_kernel(
    const float* __restrict__ q32, const float* __restrict__ nd,
    const float* __restrict__ Ssum, const float* __restrict__ emw,
    unsigned short* __restrict__ yh, int T_, int D_)
{
    size_t i = (size_t)blockIdx.x * blockDim.x + threadIdx.x;
    int d = (int)(i % D_);
    size_t bt = i / D_;
    int t = (int)(bt % T_);
    int b = (int)(bt / T_);
    const float* ndp = nd + ((size_t)b * T_ + t) * (2 * D_);
    float e = emw[t];
    float num = ndp[d]      + e * Ssum[b * 2 * D_ + d];
    float den = ndp[D_ + d] + e * Ssum[b * 2 * D_ + D_ + d];
    float q = q32[i];
    float sig = 1.0f / (1.0f + __expf(-q));
    yh[i] = cvt_bf16(sig * num / den);
}

// ---------------------------------------------------------------------------
// 10) out = LayerNorm(xa + xb)*gamma + beta; optional bf16 copy. 1 block/row.
// ---------------------------------------------------------------------------
__global__ __launch_bounds__(256) void add_ln_kernel(
    const float* __restrict__ xa, const float* __restrict__ xb,
    const float* __restrict__ gamma, const float* __restrict__ beta,
    float* __restrict__ out32, unsigned short* __restrict__ out16, int D_)
{
    int row = blockIdx.x;
    const float* pa = xa + (size_t)row * D_;
    const float* pb = xb + (size_t)row * D_;
    float v[4];
    float s = 0.0f, ss = 0.0f;
    for (int i = 0; i < 4; ++i) {
        int c = threadIdx.x + i * 256;
        float t = pa[c] + pb[c];
        v[i] = t; s += t; ss += t * t;
    }
    __shared__ float r1[256];
    __shared__ float r2[256];
    r1[threadIdx.x] = s; r2[threadIdx.x] = ss; __syncthreads();
    for (int st = 128; st > 0; st >>= 1) {
        if (threadIdx.x < st) { r1[threadIdx.x] += r1[threadIdx.x + st]; r2[threadIdx.x] += r2[threadIdx.x + st]; }
        __syncthreads();
    }
    float mu  = r1[0] / D_;
    float var = r2[0] / D_ - mu * mu;
    float inv = rsqrtf(var + 1e-5f);
    for (int i = 0; i < 4; ++i) {
        int c = threadIdx.x + i * 256;
        float val = (v[i] - mu) * inv * gamma[c] + beta[c];
        out32[(size_t)row * D_ + c] = val;
        if (out16) out16[(size_t)row * D_ + c] = cvt_bf16(val);
    }
}

// ---------------------------------------------------------------------------
// Host launcher
// ---------------------------------------------------------------------------
extern "C" void kernel_launch(void* const* d_in, const int* in_sizes, int n_in,
                              void* d_out, int out_size, void* d_ws, size_t ws_size,
                              hipStream_t stream)
{
    (void)in_sizes; (void)n_in; (void)out_size; (void)ws_size;
    const int Bb = 4, T_ = 2048, D_ = 1024, H_ = 4096, S_ = 256;
    const int KW = 544, JSHIFT = 264;          // band window: 16 + 2*(S-1)=526 -> 17 k-tiles
    const int M  = Bb * T_;                    // 8192 rows

    const float* x   = (const float*)d_in[0];
    const float* pe  = (const float*)d_in[1];
    const float* wq  = (const float*)d_in[2];
    const float* bq  = (const float*)d_in[3];
    const float* wk  = (const float*)d_in[4];
    const float* bk  = (const float*)d_in[5];
    const float* wv  = (const float*)d_in[6];
    const float* bv  = (const float*)d_in[7];
    const float* wo  = (const float*)d_in[8];
    const float* bo  = (const float*)d_in[9];
    const float* pb  = (const float*)d_in[10];
    const float* gamma = (const float*)d_in[11];
    const float* beta  = (const float*)d_in[12];
    const float* w1  = (const float*)d_in[13];
    const float* b1  = (const float*)d_in[14];
    const float* w2  = (const float*)d_in[15];
    const float* b2  = (const float*)d_in[16];

    const size_t MB = 1u << 20;
    char* ws = (char*)d_ws;
    float*          emb32 = (float*)(ws);                        //  32MB, later a32
    unsigned short* h16   = (unsigned short*)(ws + 32  * MB);    //  16MB: embh / yh / ah
    float*          s32   = (float*)(ws + 48 * MB);              //  32MB: q32 / attn32 / ff32
    float*          big   = (float*)(ws + 80 * MB);              //  64MB: k32|v32 / numden / h bf16
    float*          k32   = big;
    float*          v32   = (float*)(ws + 112 * MB);
    float*          nd    = big;
    unsigned short* hbuf  = (unsigned short*)big;
    unsigned short* ekvT  = (unsigned short*)(ws + 144 * MB);    //  32MB
    unsigned short* ewp   = (unsigned short*)(ws + 176 * MB);    // 2.2MB
    unsigned short* wqT   = (unsigned short*)(ws + 180 * MB);
    unsigned short* wkT   = (unsigned short*)(ws + 182 * MB);
    unsigned short* wvT   = (unsigned short*)(ws + 184 * MB);
    unsigned short* woT   = (unsigned short*)(ws + 186 * MB);
    unsigned short* w1T   = (unsigned short*)(ws + 188 * MB);    //   8MB [H,D]
    unsigned short* w2T   = (unsigned short*)(ws + 196 * MB);    //   8MB [D,H]
    float*          maxk  = (float*)(ws + 204 * MB);
    float*          emw   = (float*)(ws + 204 * MB + 64 * 1024);
    float*          Ssum  = (float*)(ws + 204 * MB + 128 * 1024);

    const size_t nBTD = (size_t)Bb * T_ * D_;                    // 8,388,608
    dim3 blk256(256);

    // 1) embed
    add_embed_kernel<<<(unsigned)(nBTD / 256), blk256, 0, stream>>>(x, pe, emb32, h16, nBTD);

    // 2) weight transposes (f32 [K,N] -> bf16 [N,K])
    transpose_cvt_kernel<<<dim3(D_ / 32, D_ / 32), dim3(32, 8), 0, stream>>>(wq, wqT, D_, D_);
    transpose_cvt_kernel<<<dim3(D_ / 32, D_ / 32), dim3(32, 8), 0, stream>>>(wk, wkT, D_, D_);
    transpose_cvt_kernel<<<dim3(D_ / 32, D_ / 32), dim3(32, 8), 0, stream>>>(wv, wvT, D_, D_);
    transpose_cvt_kernel<<<dim3(D_ / 32, D_ / 32), dim3(32, 8), 0, stream>>>(wo, woT, D_, D_);
    transpose_cvt_kernel<<<dim3(H_ / 32, D_ / 32), dim3(32, 8), 0, stream>>>(w1, w1T, D_, H_);
    transpose_cvt_kernel<<<dim3(D_ / 32, H_ / 32), dim3(32, 8), 0, stream>>>(w2, w2T, H_, D_);

    // 3) QKV projections (coop wmma): WG grid = (M/128)*(N/64)
    int wgQ = (M / 128) * (D_ / 64);
    gemm_bf16_coop_kernel<<<wgQ, blk256, 0, stream>>>(h16, D_, wqT, D_, s32, nullptr, D_, bq, M, D_, D_, 0);
    gemm_bf16_coop_kernel<<<wgQ, blk256, 0, stream>>>(h16, D_, wkT, D_, k32, nullptr, D_, bk, M, D_, D_, 0);
    gemm_bf16_coop_kernel<<<wgQ, blk256, 0, stream>>>(h16, D_, wvT, D_, v32, nullptr, D_, bv, M, D_, D_, 0);

    // 4) max over time of k
    colmax_kernel<<<(Bb * D_ + 255) / 256, blk256, 0, stream>>>(k32, maxk, Bb, T_, D_);

    // 5) build transposed [ek*v ; ek] bf16
    build_ekvT_kernel<<<dim3(T_ / 32, D_ / 32, Bb), dim3(32, 8), 0, stream>>>(k32, v32, maxk, ekvT, T_, D_);

    // 6) global column sums
    colsum_kernel<<<Bb * 2 * D_, blk256, 0, stream>>>(ekvT, Ssum, T_);

    // 7) banded ew matrix (and exp(-max_w))
    ew_pack_kernel<<<T_ / 16, blk256, 0, stream>>>(pb, ewp, emw, T_, S_, KW, JSHIFT);

    // 8) banded GEMM -> num||den (wmma)
    int wavesB = Bb * (T_ / 16) * ((2 * D_) / 64);
    band_gemm_kernel<<<(wavesB + 7) / 8, blk256, 0, stream>>>(ewp, ekvT, nd, T_, 2 * D_, KW, JSHIFT);

    // 9) combine -> y (bf16)
    combine_kernel<<<(unsigned)(nBTD / 256), blk256, 0, stream>>>(s32, nd, Ssum, emw, h16, T_, D_);

    // 10) output projection: attn = y @ wo + bo -> s32
    gemm_bf16_coop_kernel<<<wgQ, blk256, 0, stream>>>(h16, D_, woT, D_, s32, nullptr, D_, bo, M, D_, D_, 0);

    // 11) a = LN(attn + emb)  (a32 overwrites emb32; ah -> h16)
    add_ln_kernel<<<M, blk256, 0, stream>>>(s32, emb32, gamma, beta, emb32, h16, D_);

    // 12) FFN up + relu -> bf16 h
    int wgF1 = (M / 128) * (H_ / 64);
    gemm_bf16_coop_kernel<<<wgF1, blk256, 0, stream>>>(h16, D_, w1T, D_, nullptr, hbuf, H_, b1, M, H_, D_, 1);

    // 13) FFN down -> ff (f32, s32)
    gemm_bf16_coop_kernel<<<wgQ, blk256, 0, stream>>>(hbuf, H_, w2T, H_, s32, nullptr, D_, b2, M, D_, H_, 0);

    // 14) out = LN(ff + a)
    add_ln_kernel<<<M, blk256, 0, stream>>>(s32, emb32, gamma, beta, (float*)d_out, nullptr, D_);
}
